// PSLoraLinear_85804856639978
// MI455X (gfx1250) — matrix operand order repaired and checked by reading, not compiled, not run on hardware
//
#include <hip/hip_runtime.h>

// ---------------------------------------------------------------------------
// PSLoraLinear for MI455X (gfx1250, wave32, WMMA):
//   out = x @ W^T + b + 2.0 * (einsum('bi,bir->br', x, A[idx]) @ Bw^T)
// B=2048, IN=4096, OUT=4096, R=16, L=64.
// GEMM is compute-bound at f32 -> convert once to bf16 and run on
// v_wmma_f32_16x16x32_bf16. K-loop feeds LDS via the CDNA5 async
// global->LDS path (ASYNCcnt) with double buffering when available.
// ---------------------------------------------------------------------------

typedef __attribute__((ext_vector_type(16))) __bf16 v16bf;
typedef __attribute__((ext_vector_type(8)))  __bf16 v8bf;
typedef __attribute__((ext_vector_type(8)))  float  v8f;
typedef __attribute__((ext_vector_type(4)))  int    v4i;

#define B_SZ    2048
#define IN_SZ   4096
#define OUT_SZ  4096
#define R_SZ    16
#define LORA_SCALING 2.0f

#define TM 128          // block tile M
#define TN 128          // block tile N
#define TK 32           // K step (bf16 WMMA K)
#define PITCH 40        // LDS row pitch in bf16 elems (32 + 8 pad, 80B, 16B-aligned)
#define ABUF (TM * PITCH)
#define BBUF (TN * PITCH)

// CDNA5 async global->LDS copy (GLOBAL_LOAD_ASYNC_TO_LDS_B128, ASYNCcnt).
// Guarded: host pass and toolchains without the builtin take the staged path.
#if defined(__gfx1250__) && \
    __has_builtin(__builtin_amdgcn_global_load_async_to_lds_b128) && \
    __has_builtin(__builtin_amdgcn_s_wait_asynccnt)
#define USE_ASYNC_LDS 1
// builtin signature (from hipcc diagnostic): (v4i addrspace(1)*, v4i addrspace(3)*, imm, imm)
typedef __attribute__((address_space(1))) v4i gv4i;
typedef __attribute__((address_space(3))) v4i lv4i;
#else
#define USE_ASYNC_LDS 0
#endif

// ---------------- f32 -> bf16 bulk convert (8 elems / thread) ---------------
__global__ void cvt_bf16_kernel(const float* __restrict__ src,
                                __bf16* __restrict__ dst, int n8) {
  int i = blockIdx.x * blockDim.x + threadIdx.x;
  if (i >= n8) return;
  const float4* s = (const float4*)src + (size_t)i * 2;
  float4 a = s[0], b = s[1];
  v8bf o;
  o[0] = (__bf16)a.x; o[1] = (__bf16)a.y; o[2] = (__bf16)a.z; o[3] = (__bf16)a.w;
  o[4] = (__bf16)b.x; o[5] = (__bf16)b.y; o[6] = (__bf16)b.z; o[7] = (__bf16)b.w;
  *((v8bf*)dst + i) = o;
}

// ------------- z[b,r] = sum_i x[b,i] * A[idx[b], i, r]  (L2-resident) -------
__global__ void lora_z_kernel(const float* __restrict__ x,
                              const int* __restrict__ lidx,
                              const float* __restrict__ loraA,
                              float* __restrict__ z) {
  const int b = blockIdx.x;
  const int t = threadIdx.x;
  const int idx = lidx[b];
  const float* __restrict__ A  = loraA + (size_t)idx * (IN_SZ * R_SZ);
  const float* __restrict__ xb = x + (size_t)b * IN_SZ;

  float acc[R_SZ];
#pragma unroll
  for (int r = 0; r < R_SZ; ++r) acc[r] = 0.0f;

  for (int i = t; i < IN_SZ; i += 256) {
    float xv = xb[i];
    const float4* Ar = (const float4*)(A + (size_t)i * R_SZ);
    float4 a0 = Ar[0], a1 = Ar[1], a2 = Ar[2], a3 = Ar[3];
    acc[0]  += xv * a0.x; acc[1]  += xv * a0.y; acc[2]  += xv * a0.z; acc[3]  += xv * a0.w;
    acc[4]  += xv * a1.x; acc[5]  += xv * a1.y; acc[6]  += xv * a1.z; acc[7]  += xv * a1.w;
    acc[8]  += xv * a2.x; acc[9]  += xv * a2.y; acc[10] += xv * a2.z; acc[11] += xv * a2.w;
    acc[12] += xv * a3.x; acc[13] += xv * a3.y; acc[14] += xv * a3.z; acc[15] += xv * a3.w;
  }

  // wave32 reduce, then cross-wave via LDS
#pragma unroll
  for (int r = 0; r < R_SZ; ++r)
#pragma unroll
    for (int off = 16; off > 0; off >>= 1)
      acc[r] += __shfl_down(acc[r], off, 32);

  __shared__ float red[8][R_SZ];
  const int lane = t & 31, wv = t >> 5;
  if (lane == 0) {
#pragma unroll
    for (int r = 0; r < R_SZ; ++r) red[wv][r] = acc[r];
  }
  __syncthreads();
  if (t < R_SZ) {
    float s = 0.0f;
#pragma unroll
    for (int w = 0; w < 8; ++w) s += red[w][t];
    z[(size_t)b * R_SZ + t] = s;
  }
}

// -------- helpers for WMMA fragment assembly from LDS (documented layouts) --
__device__ __forceinline__ v16bf concat16(v8bf lo, v8bf hi) {
  return __builtin_shufflevector(lo, hi, 0, 1, 2, 3, 4, 5, 6, 7,
                                         8, 9, 10, 11, 12, 13, 14, 15);
}

__device__ __forceinline__ v8bf pack8(float4 a, float4 b, float s) {
  v8bf o;
  o[0] = (__bf16)(a.x * s); o[1] = (__bf16)(a.y * s);
  o[2] = (__bf16)(a.z * s); o[3] = (__bf16)(a.w * s);
  o[4] = (__bf16)(b.x * s); o[5] = (__bf16)(b.y * s);
  o[6] = (__bf16)(b.z * s); o[7] = (__bf16)(b.w * s);
  return o;
}

__device__ __forceinline__ void async_cp_b128(const __bf16* g, __bf16* s) {
#if USE_ASYNC_LDS
  __builtin_amdgcn_global_load_async_to_lds_b128(
      (gv4i*)g, (lv4i*)s, /*offset=*/0, /*cpol=*/0);
#else
  (void)g; (void)s;
#endif
}

// One K-step of WMMA work for this wave: 4 A-frags x 2 B-frags -> 8 wmma.
// A 16x32 bf16 layout: lanes 0-15: M=lane,  K 0-7 / 16-23;
//                      lanes 16-31: M=lane-16, K 8-15 / 24-31.
// B 32x16 bf16 layout: lanes 0-15: N=lane,  K 0-7 / 16-23;
//                      lanes 16-31: N=lane-16, K 8-15 / 24-31.
__device__ __forceinline__ void wmma_tile_step(const __bf16* a_lds,
                                               const __bf16* b_lds,
                                               int waveM, int waveN,
                                               int lrow, int lhi,
                                               v8f (&acc)[4][2]) {
  v16bf afr[4];
#pragma unroll
  for (int mi = 0; mi < 4; ++mi) {
    const int row = waveM * 64 + mi * 16 + lrow;
    v8bf lo = *(const v8bf*)(a_lds + row * PITCH + lhi * 8);
    v8bf hi = *(const v8bf*)(a_lds + row * PITCH + 16 + lhi * 8);
    afr[mi] = concat16(lo, hi);
  }
#pragma unroll
  for (int nj = 0; nj < 2; ++nj) {
    const int col = waveN * 32 + nj * 16 + lrow;
    const int kb  = lhi * 8;
    v8bf lo = *(const v8bf*)(b_lds + col * PITCH + kb);
    v8bf hi = *(const v8bf*)(b_lds + col * PITCH + kb + 16);
    v16bf bfr = concat16(lo, hi);
#pragma unroll
    for (int mi = 0; mi < 4; ++mi)
      acc[mi][nj] = __builtin_amdgcn_wmma_f32_16x16x32_bf16(
          false, afr[mi], false, bfr, (short)0, acc[mi][nj], false, false);
  }
}

// ----------------------------- main WMMA GEMM -------------------------------
// out[m,n] = sum_k xb[m,k]*wb[n,k]  + bias[n] + sum_r zb[m,r]*(2*Bw[n,r])
__global__ __launch_bounds__(256, 1)
void lora_gemm_kernel(const __bf16* __restrict__ Xb,
                      const __bf16* __restrict__ Wb,
                      const float* __restrict__ bias,
                      const float* __restrict__ z,
                      const float* __restrict__ Bw,
                      float* __restrict__ out) {
  __shared__ __align__(16) __bf16 As[2 * ABUF];   // 20 KB (double buffer)
  __shared__ __align__(16) __bf16 Bs[2 * BBUF];   // 20 KB

  const int t     = threadIdx.x;
  const int lane  = t & 31;
  const int wv    = t >> 5;          // 0..7 waves
  const int waveM = wv >> 2;         // 0..1 -> 64 rows each
  const int waveN = wv & 3;          // 0..3 -> 32 cols each
  const int m0    = blockIdx.y * TM;
  const int n0    = blockIdx.x * TN;

  const int lrow = lane & 15;        // row/col within 16x16 tile
  const int lhi  = lane >> 4;        // half-wave selector

  v8f acc[4][2];
#pragma unroll
  for (int mi = 0; mi < 4; ++mi)
#pragma unroll
    for (int nj = 0; nj < 2; ++nj)
#pragma unroll
      for (int e = 0; e < 8; ++e) acc[mi][nj][e] = 0.0f;

  // cooperative tile loading: 256 threads x 32B cover 128 rows x 64B
  const int    ldRow  = t >> 1;
  const int    ldHalf = (t & 1) * 16;   // bf16-element offset (32B)
  const size_t gaBase = (size_t)(m0 + ldRow) * IN_SZ + ldHalf;
  const size_t gbBase = (size_t)(n0 + ldRow) * IN_SZ + ldHalf;
  const int    sOff   = ldRow * PITCH + ldHalf;
  const int    nTiles = IN_SZ / TK;     // 128

#if USE_ASYNC_LDS
  // ---- pipelined: async global->LDS (ASYNCcnt), double-buffered -----------
  {
    async_cp_b128(Xb + gaBase,     As + sOff);
    async_cp_b128(Xb + gaBase + 8, As + sOff + 8);
    async_cp_b128(Wb + gbBase,     Bs + sOff);
    async_cp_b128(Wb + gbBase + 8, Bs + sOff + 8);
  }
  for (int kt = 0; kt < nTiles; ++kt) {
    __builtin_amdgcn_s_wait_asynccnt(0);   // our portion of buf[kt&1] landed
    __syncthreads();                       // everyone's portion visible
    const int cur = kt & 1;
    if (kt + 1 < nTiles) {                 // stream tile kt+1 into other buf
      const size_t ka = (size_t)(kt + 1) * TK;
      const int nxt = cur ^ 1;
      async_cp_b128(Xb + gaBase + ka,     As + nxt * ABUF + sOff);
      async_cp_b128(Xb + gaBase + ka + 8, As + nxt * ABUF + sOff + 8);
      async_cp_b128(Wb + gbBase + ka,     Bs + nxt * BBUF + sOff);
      async_cp_b128(Wb + gbBase + ka + 8, Bs + nxt * BBUF + sOff + 8);
      if (kt + 2 < nTiles) {               // keep L2 warm one tile further out
        __builtin_prefetch((const void*)(Xb + gaBase + ka + TK), 0, 3);
        __builtin_prefetch((const void*)(Wb + gbBase + ka + TK), 0, 3);
      }
    }
    wmma_tile_step(As + cur * ABUF, Bs + cur * BBUF, waveM, waveN, lrow, lhi, acc);
  }
#else
  // ---- fallback: register-staged tiles, single buffer ---------------------
  for (int k0 = 0; k0 < IN_SZ; k0 += TK) {
    const uint4* ga = (const uint4*)(Xb + gaBase + k0);
    const uint4* gb = (const uint4*)(Wb + gbBase + k0);
    uint4 ax0 = ga[0], ax1 = ga[1];
    uint4 bx0 = gb[0], bx1 = gb[1];
    if (k0 + TK < IN_SZ) {
      __builtin_prefetch((const void*)(Xb + gaBase + k0 + TK), 0, 3);
      __builtin_prefetch((const void*)(Wb + gbBase + k0 + TK), 0, 3);
    }
    __syncthreads();
    uint4* sa = (uint4*)(As + sOff);
    uint4* sb = (uint4*)(Bs + sOff);
    sa[0] = ax0; sa[1] = ax1;
    sb[0] = bx0; sb[1] = bx1;
    __syncthreads();
    wmma_tile_step(As, Bs, waveM, waveN, lrow, lhi, acc);
  }
#endif

  // ---- LoRA epilogue as one more WMMA: K=32 zero-padded bf16 tiles --------
  __syncthreads();
  {
    const uint4 zz = make_uint4(0u, 0u, 0u, 0u);
    if (t < 128) {                   // z tile -> As buf0 (128 x 16, pad K 16..31)
      const float4* zr = (const float4*)(z + (size_t)(m0 + t) * R_SZ);
      float4 z0 = zr[0], z1 = zr[1], z2 = zr[2], z3 = zr[3];
      *(v8bf*)(As + t * PITCH + 0)   = pack8(z0, z1, 1.0f);
      *(v8bf*)(As + t * PITCH + 8)   = pack8(z2, z3, 1.0f);
      *(uint4*)(As + t * PITCH + 16) = zz;
      *(uint4*)(As + t * PITCH + 24) = zz;
    } else {                         // SCALING*Bw tile -> Bs buf0 (128 x 16, pad)
      const int r = t - 128;
      const float4* br = (const float4*)(Bw + (size_t)(n0 + r) * R_SZ);
      float4 b0 = br[0], b1 = br[1], b2 = br[2], b3 = br[3];
      *(v8bf*)(Bs + r * PITCH + 0)   = pack8(b0, b1, LORA_SCALING);
      *(v8bf*)(Bs + r * PITCH + 8)   = pack8(b2, b3, LORA_SCALING);
      *(uint4*)(Bs + r * PITCH + 16) = zz;
      *(uint4*)(Bs + r * PITCH + 24) = zz;
    }
  }
  __syncthreads();
  wmma_tile_step(As, Bs, waveM, waveN, lrow, lhi, acc);

  // ---- bias + store: C/D layout: lane 0-15 -> N=lane, M=v; 16-31 -> M=v+8 -
#pragma unroll
  for (int nj = 0; nj < 2; ++nj) {
    const int n = n0 + waveN * 32 + nj * 16 + lrow;
    const float bn = bias[n];
#pragma unroll
    for (int mi = 0; mi < 4; ++mi) {
      const int mBase = m0 + waveM * 64 + mi * 16 + lhi * 8;
#pragma unroll
      for (int v = 0; v < 8; ++v)
        out[(size_t)(mBase + v) * OUT_SZ + n] = acc[mi][nj][v] + bn;
    }
  }
}

// ---------------------------------------------------------------------------
extern "C" void kernel_launch(void* const* d_in, const int* in_sizes, int n_in,
                              void* d_out, int out_size, void* d_ws, size_t ws_size,
                              hipStream_t stream) {
  const float* x      = (const float*)d_in[0];
  const int*   lidx   = (const int*)d_in[1];
  const float* baseW  = (const float*)d_in[2];
  const float* baseb  = (const float*)d_in[3];
  const float* loraA  = (const float*)d_in[4];
  const float* loraBw = (const float*)d_in[5];
  float* out = (float*)d_out;

  // workspace: x_bf16 (16MB) | W_bf16 (32MB) | z (128KB)
  char* ws = (char*)d_ws;
  __bf16* xb = (__bf16*)ws;
  __bf16* wb = (__bf16*)(ws + (size_t)B_SZ * IN_SZ * sizeof(__bf16));
  float*  z  = (float*)(ws + (size_t)B_SZ * IN_SZ * sizeof(__bf16)
                           + (size_t)OUT_SZ * IN_SZ * sizeof(__bf16));

  {
    const int n8 = (B_SZ * IN_SZ) / 8;
    cvt_bf16_kernel<<<(n8 + 255) / 256, 256, 0, stream>>>(x, xb, n8);
  }
  {
    const int n8 = (OUT_SZ * IN_SZ) / 8;
    cvt_bf16_kernel<<<(n8 + 255) / 256, 256, 0, stream>>>(baseW, wb, n8);
  }
  lora_z_kernel<<<B_SZ, 256, 0, stream>>>(x, lidx, loraA, z);

  dim3 grid(OUT_SZ / TN, B_SZ / TM);   // 32 x 16 = 512 blocks, 8 waves each
  lora_gemm_kernel<<<grid, 256, 0, stream>>>(xb, wb, baseb, z, loraBw, out);
}
